// InteractionPredictor_49022756716913
// MI455X (gfx1250) — matrix operand order repaired and checked by reading, not compiled, not run on hardware
//
#include <hip/hip_runtime.h>
#include <math.h>

#define SQRT3F     1.7320508075688772f
#define INV_SQRT3F 0.5773502691896258f

typedef __attribute__((ext_vector_type(16))) _Float16 v16h;
typedef __attribute__((ext_vector_type(8)))  _Float16 v8h;
typedef __attribute__((ext_vector_type(8)))  float    v8f;

__device__ __forceinline__ float silu_f(float v) { return v / (1.0f + __expf(-v)); }

// ---------------------------------------------------------------- utilities
__global__ void k_zero(float* __restrict__ p, int n) {
  int i = blockIdx.x * blockDim.x + threadIdx.x;
  if (i < n) p[i] = 0.0f;
}
__global__ void k_zero_h(_Float16* __restrict__ p, int n) {
  int i = blockIdx.x * blockDim.x + threadIdx.x;
  if (i < n) p[i] = (_Float16)0.0f;
}

// Pack B[K,N] (f32) into WMMA-fragment-friendly f16: dst[kt][n][kk] = B[kt*32+kk, n]
// transpose=1: B[k,n] = src[n*Kvalid + k]  (used for W2_inter^T)
__global__ void k_pack_b(const float* __restrict__ src, _Float16* __restrict__ dst,
                         int Kvalid, int Nvalid, int Npad, int total, int transpose) {
  int idx = blockIdx.x * blockDim.x + threadIdx.x;
  if (idx >= total) return;
  int kk = idx & 31;
  int n  = (idx >> 5) % Npad;
  int kt = idx / (Npad << 5);
  int k  = (kt << 5) + kk;
  float v = 0.0f;
  if (k < Kvalid && n < Nvalid)
    v = transpose ? src[(size_t)n * Kvalid + k] : src[(size_t)k * Nvalid + n];
  dst[(((size_t)kt * Npad + n) << 5) + kk] = (_Float16)v;
}

// First MLP layer (K<=10): H[r,k] = act(sum_u X[r,u]*W1[u,k]*scale)
__global__ void k_row_mlp64(const float* __restrict__ X, int M, int K,
                            const float* __restrict__ W1, float scale, int act,
                            _Float16* __restrict__ H) {
  int r = blockIdx.x * blockDim.x + threadIdx.x;
  if (r >= M) return;
  float xin[10];
  for (int u = 0; u < K; ++u) xin[u] = X[(size_t)r * K + u];
  for (int k = 0; k < 64; ++k) {
    float acc = 0.0f;
    for (int u = 0; u < K; ++u) acc += xin[u] * W1[u * 64 + k];
    acc *= scale;
    if (act) acc = silu_f(acc);
    H[(size_t)r * 64 + k] = (_Float16)acc;
  }
}

// ---------------------------------------------------------------- WMMA GEMM
// C[M,Npad] = scale * A[M,Kpad](f16,row-major) @ Bpack.
// One wave computes a 32x(NTILES*16) tile; A fragments reused NTILES x, B
// fragments 2x. NTILES and the K-tile count are compile-time -> the inner
// loop is branch-free and fully unrolled for software pipelining.
// M must be a multiple of 32 (caller pads A/C buffers).
template <typename OutT, int NTILES, int NKT>
__global__ void k_gemm_wmma(const _Float16* __restrict__ A,
                            const _Float16* __restrict__ B,
                            OutT* __restrict__ C,
                            int Kpad, int Npad, int ntBase, float scale) {
  int mt = blockIdx.x;            // 32-row tile
  int nt = ntBase + blockIdx.y;   // 64-col tile index (16-col subtiles = 4*nt+j)
  int lane = threadIdx.x & 31;
  int hl = lane >> 4;
  const _Float16* arow0 = A + (size_t)((mt << 5) + (lane & 15)) * Kpad;
  const _Float16* arow1 = arow0 + (size_t)16 * Kpad;
  const int nbase = (nt << 6) + (lane & 15);
  v8f acc0[NTILES] = {}, acc1[NTILES] = {};
#pragma unroll
  for (int kt = 0; kt < NKT; ++kt) {
    v8h p0 = *(const v8h*)(arow0 + kt * 32 + 16 * hl);
    v8h p1 = *(const v8h*)(arow0 + kt * 32 + 16 * hl + 8);
    v8h q0 = *(const v8h*)(arow1 + kt * 32 + 16 * hl);
    v8h q1 = *(const v8h*)(arow1 + kt * 32 + 16 * hl + 8);
    v16h a0, a1;
#pragma unroll
    for (int i = 0; i < 8; ++i) {
      a0[i] = p0[i]; a0[i + 8] = p1[i];
      a1[i] = q0[i]; a1[i + 8] = q1[i];
    }
#pragma unroll
    for (int j = 0; j < NTILES; ++j) {
      v16h b = *(const v16h*)(B + (((size_t)kt * Npad + nbase + j * 16) << 5) +
                              16 * hl);
      acc0[j] = __builtin_amdgcn_wmma_f32_16x16x32_f16(false, a0, false, b,
                                                       (short)0, acc0[j],
                                                       false, false);
      acc1[j] = __builtin_amdgcn_wmma_f32_16x16x32_f16(false, a1, false, b,
                                                       (short)0, acc1[j],
                                                       false, false);
    }
  }
#pragma unroll
  for (int j = 0; j < NTILES; ++j) {
    int cn = nbase + j * 16;
#pragma unroll
    for (int r = 0; r < 8; ++r) {
      int cm = (mt << 5) + r + 8 * hl;
      C[(size_t)cm * Npad + cn] = (OutT)(acc0[j][r] * scale);
      C[(size_t)(cm + 16) * Npad + cn] = (OutT)(acc1[j][r] * scale);
    }
  }
}

// host-side dispatch: full 64-col tiles with NTILES=4, one remainder launch.
template <typename OutT>
static void launch_gemm(const _Float16* A, const _Float16* B, OutT* C,
                        int Mg, int Kpad, int Npad, float scale,
                        hipStream_t stream) {
  int t16 = Npad >> 4, full = t16 >> 2, rem = t16 & 3;
  if (full) {
    dim3 g(Mg / 32, full);
    if (Kpad == 64)
      k_gemm_wmma<OutT, 4, 2><<<g, 32, 0, stream>>>(A, B, C, Kpad, Npad, 0, scale);
    else
      k_gemm_wmma<OutT, 4, 10><<<g, 32, 0, stream>>>(A, B, C, Kpad, Npad, 0, scale);
  }
  if (rem) {
    dim3 g(Mg / 32, 1);
    if (Kpad == 64) {
      if (rem == 1)      k_gemm_wmma<OutT, 1, 2><<<g, 32, 0, stream>>>(A, B, C, Kpad, Npad, full, scale);
      else if (rem == 2) k_gemm_wmma<OutT, 2, 2><<<g, 32, 0, stream>>>(A, B, C, Kpad, Npad, full, scale);
      else               k_gemm_wmma<OutT, 3, 2><<<g, 32, 0, stream>>>(A, B, C, Kpad, Npad, full, scale);
    } else {
      if (rem == 1)      k_gemm_wmma<OutT, 1, 10><<<g, 32, 0, stream>>>(A, B, C, Kpad, Npad, full, scale);
      else if (rem == 2) k_gemm_wmma<OutT, 2, 10><<<g, 32, 0, stream>>>(A, B, C, Kpad, Npad, full, scale);
      else               k_gemm_wmma<OutT, 3, 10><<<g, 32, 0, stream>>>(A, B, C, Kpad, Npad, full, scale);
    }
  }
}

// ------------------------------------------- message FCTP + scatter (epilogue)
// 8 threads per edge (one per scalar output w); lanes w<2 also handle vectors.
template <bool HASV>
__global__ void k_msg_scatter(const int* __restrict__ ei, int E,
                              const float* __restrict__ pos,
                              const float* __restrict__ ns,
                              const float* __restrict__ nv,
                              const _Float16* __restrict__ wm, int ldw,
                              float* __restrict__ agg_s, float* __restrict__ agg_v,
                              float rs, float rv, float deg) {
  int idx = blockIdx.x * blockDim.x + threadIdx.x;
  if (idx >= E * 8) return;
  int e = idx >> 3, w = idx & 7;
  int src = ei[e], dst = ei[E + e];
  float vx = pos[dst * 3 + 0] - pos[src * 3 + 0];
  float vy = pos[dst * 3 + 1] - pos[src * 3 + 1];
  float vz = pos[dst * 3 + 2] - pos[src * 3 + 2];
  float invn = rsqrtf(vx * vx + vy * vy + vz * vz + 1e-12f);
  float shx = SQRT3F * vx * invn, shy = SQRT3F * vy * invn, shz = SQRT3F * vz * invn;
  const _Float16* W = wm + (size_t)e * ldw;
  float s[16];
  for (int u = 0; u < 16; ++u) s[u] = ns[src * 16 + u];
  // scalar output: Wss @0 (16,1,8); HASV adds Wvv @128 (4,1,8)
  float ms = 0.0f;
  for (int u = 0; u < 16; ++u) ms += s[u] * (float)W[u * 8 + w];
  if (HASV) {
    for (int u = 0; u < 4; ++u) {
      float dv = nv[src * 12 + u * 3 + 0] * shx + nv[src * 12 + u * 3 + 1] * shy +
                 nv[src * 12 + u * 3 + 2] * shz;
      ms += dv * INV_SQRT3F * (float)W[128 + u * 8 + w];
    }
  }
  atomicAdd(&agg_s[dst * 8 + w], ms * rs * deg);
  if (w < 2) {
    const int o_sv = HASV ? 160 : 128;  // Wsv (16,1,2)
    float t = 0.0f;
    for (int u = 0; u < 16; ++u) t += s[u] * (float)W[o_sv + u * 2 + w];
    float mvx = t * shx, mvy = t * shy, mvz = t * shz;
    if (HASV) {  // Wvs @192 (4,1,2)
      for (int u = 0; u < 4; ++u) {
        float wv = (float)W[192 + u * 2 + w];
        mvx += nv[src * 12 + u * 3 + 0] * wv;
        mvy += nv[src * 12 + u * 3 + 1] * wv;
        mvz += nv[src * 12 + u * 3 + 2] * wv;
      }
    }
    atomicAdd(&agg_v[dst * 6 + w * 3 + 0], mvx * rv * deg);
    atomicAdd(&agg_v[dst * 6 + w * 3 + 1], mvy * rv * deg);
    atomicAdd(&agg_v[dst * 6 + w * 3 + 2], mvz * rv * deg);
  }
}

// ------------------------------------------- node update FCTP (epilogue)
// 16 threads per node (one per scalar output w); lanes w<4 also do vectors.
template <bool HASV>
__global__ void k_upd_fctp(int base, int Mc,
                           const float* __restrict__ ns, const float* __restrict__ nv,
                           const float* __restrict__ agg_s,
                           const float* __restrict__ agg_v,
                           const _Float16* __restrict__ wu, int ldw,
                           float* __restrict__ ns_out, float* __restrict__ nv_out,
                           float rs, float rv) {
  int idx = blockIdx.x * blockDim.x + threadIdx.x;
  if (idx >= Mc * 16) return;
  int n = idx >> 4, w = idx & 15;
  int g = base + n;
  const _Float16* W = wu + (size_t)n * ldw;
  float s[16], as[8], av[2][3];
  for (int u = 0; u < 16; ++u) s[u] = ns[g * 16 + u];
  for (int v = 0; v < 8; ++v) as[v] = agg_s[g * 8 + v];
  for (int v = 0; v < 2; ++v)
    for (int i = 0; i < 3; ++i) av[v][i] = agg_v[g * 6 + v * 3 + i];
  // scalar: Wss @0 (16,8,16); HASV adds Wvv @2048 (4,2,16)
  float acc = 0.0f;
  for (int u = 0; u < 16; ++u)
    for (int v = 0; v < 8; ++v) acc += s[u] * as[v] * (float)W[(u * 8 + v) * 16 + w];
  if (HASV) {
    for (int u = 0; u < 4; ++u) {
      float ux = nv[g * 12 + u * 3 + 0], uy = nv[g * 12 + u * 3 + 1],
            uz = nv[g * 12 + u * 3 + 2];
      for (int v = 0; v < 2; ++v) {
        float dv = ux * av[v][0] + uy * av[v][1] + uz * av[v][2];
        acc += dv * INV_SQRT3F * (float)W[2048 + (u * 2 + v) * 16 + w];
      }
    }
  }
  ns_out[(size_t)g * 16 + w] = acc * rs;
  if (w < 4) {
    const int o_sv = HASV ? 2176 : 2048;  // Wsv (16,2,4)
    float ox = 0.0f, oy = 0.0f, oz = 0.0f;
    for (int u = 0; u < 16; ++u)
      for (int v = 0; v < 2; ++v) {
        float c = s[u] * (float)W[o_sv + (u * 2 + v) * 4 + w];
        ox += c * av[v][0]; oy += c * av[v][1]; oz += c * av[v][2];
      }
    if (HASV) {  // Wvs @2304 (4,8,4)
      for (int u = 0; u < 4; ++u) {
        float ux = nv[g * 12 + u * 3 + 0], uy = nv[g * 12 + u * 3 + 1],
              uz = nv[g * 12 + u * 3 + 2];
        for (int v = 0; v < 8; ++v) {
          float c = as[v] * (float)W[2304 + (u * 8 + v) * 4 + w];
          ox += c * ux; oy += c * uy; oz += c * uz;
        }
      }
    }
    nv_out[(size_t)g * 12 + w * 3 + 0] = ox * rv;
    nv_out[(size_t)g * 12 + w * 3 + 1] = oy * rv;
    nv_out[(size_t)g * 12 + w * 3 + 2] = oz * rv;
  }
}

// ------------------------------------------- interaction: build f + silu(h)
// one wave per inter-edge
__global__ void k_inter_build(const int* __restrict__ iei, int Ei, int base, int Mc,
                              const float* __restrict__ pos,
                              const float* __restrict__ ns,
                              const float* __restrict__ nv,
                              const float* __restrict__ W1,  // [25,64]
                              _Float16* __restrict__ F,      // [Mc,320]
                              _Float16* __restrict__ SH) {   // [Mc,64]
  int widx = (blockIdx.x * blockDim.x + threadIdx.x) >> 5;
  int lane = threadIdx.x & 31;
  if (widx >= Mc) return;
  int e = base + widx;
  int rec = iei[e], lig = iei[Ei + e];
  float vx = pos[lig * 3 + 0] - pos[rec * 3 + 0];
  float vy = pos[lig * 3 + 1] - pos[rec * 3 + 1];
  float vz = pos[lig * 3 + 2] - pos[rec * 3 + 2];
  float dn = sqrtf(vx * vx + vy * vy + vz * vz + 1e-12f);
  float inv = 1.0f / dn;
  float shx = SQRT3F * vx * inv, shy = SQRT3F * vy * inv, shz = SQRT3F * vz * inv;
  _Float16* Fr = F + (size_t)widx * 320;
  for (int j = lane; j < 320; j += 32) {
    float val = 0.0f;
    if (j < 272) {                       // Wss block: f[u*16+v] = in1s_u * rec_s_v
      int u = j >> 4, v = j & 15;
      float su = (u < 16) ? ns[lig * 16 + u] : 1.0f;
      val = su * ns[rec * 16 + v];
    } else if (j < 292) {                // Wvv block: dot(in1v_u, rec_v_v)/sqrt(3)
      int t = j - 272;
      int u = t >> 2, v = t & 3;
      float ax, ay, az;
      if (u < 4) { ax = nv[lig * 12 + u * 3 + 0]; ay = nv[lig * 12 + u * 3 + 1];
                   az = nv[lig * 12 + u * 3 + 2]; }
      else       { ax = shx; ay = shy; az = shz; }
      val = (ax * nv[rec * 12 + v * 3 + 0] + ay * nv[rec * 12 + v * 3 + 1] +
             az * nv[rec * 12 + v * 3 + 2]) * INV_SQRT3F;
    }
    Fr[j] = (_Float16)val;
  }
  // smooth_finite basis in registers, shared via shuffles
  float b = 0.0f;
  if (lane < 25) {
    float q = dn * (26.0f / 5.0f);
    float t1 = q - (float)lane;
    float t2 = (float)(lane + 2) - q;
    if (t1 > 0.0f && t2 > 0.0f) {
      const float C0 = 1.14136f * 7.389056099f * 5.0f;  // 1.14136*e^2*sqrt(25)
      b = C0 * __expf(-1.0f / t1) * __expf(-1.0f / t2);
    }
  }
  for (int k = lane; k < 64; k += 32) {
    float acc = 0.0f;
    for (int u = 0; u < 25; ++u) {
      float bu = __shfl(b, u);
      acc += bu * W1[u * 64 + k];
    }
    SH[(size_t)widx * 64 + k] = (_Float16)silu_f(acc * 0.2f);  // /sqrt(25), silu
  }
}

// out[e] = dot64(silu(h), T)   (scales folded into T GEMM); wave per edge
__global__ void k_inter_out(const _Float16* __restrict__ SH,
                            const float* __restrict__ T,
                            float* __restrict__ out, int Mc, int base) {
  int widx = (blockIdx.x * blockDim.x + threadIdx.x) >> 5;
  int lane = threadIdx.x & 31;
  if (widx >= Mc) return;
  size_t o = (size_t)widx * 64;
  float acc = (float)SH[o + lane] * T[o + lane] +
              (float)SH[o + lane + 32] * T[o + lane + 32];
  for (int off = 16; off > 0; off >>= 1) acc += __shfl_xor(acc, off);
  if (lane == 0) out[base + widx] = acc;
}

// ================================================================= host side
extern "C" void kernel_launch(void* const* d_in, const int* in_sizes, int n_in,
                              void* d_out, int out_size, void* d_ws, size_t ws_size,
                              hipStream_t stream) {
  const float* x   = (const float*)d_in[0];
  const float* pos = (const float*)d_in[1];
  const int*   ei  = (const int*)d_in[2];
  const float* ea  = (const float*)d_in[3];
  const int*   iei = (const int*)d_in[4];
  const int N  = in_sizes[0] / 10;
  const int E  = in_sizes[2] / 2;
  const int Ei = in_sizes[4] / 2;
  const int N32 = (N + 31) & ~31;   // GEMM-padded node count
  const int E32 = (E + 31) & ~31;

  // identify weight arrays by unique W2 element counts (robust to pytree order)
  const float *embW1 = 0, *embW2 = 0, *interW1 = 0, *interW2 = 0;
  const float *msg1W1 = 0, *msg1W2 = 0, *upd1W1 = 0, *upd1W2 = 0;
  const float *msg2W1 = 0, *msg2W2 = 0, *upd2W1 = 0, *upd2W2 = 0;
  for (int i = 5; i < n_in; ++i) {
    switch (in_sizes[i]) {
      case 64 * 16:   embW2  = (const float*)d_in[i]; embW1  = (const float*)d_in[i-1]; break;
      case 64 * 292:  interW2= (const float*)d_in[i]; interW1= (const float*)d_in[i-1]; break;
      case 64 * 160:  msg1W2 = (const float*)d_in[i]; msg1W1 = (const float*)d_in[i-1]; break;
      case 64 * 2176: upd1W2 = (const float*)d_in[i]; upd1W1 = (const float*)d_in[i-1]; break;
      case 64 * 200:  msg2W2 = (const float*)d_in[i]; msg2W1 = (const float*)d_in[i-1]; break;
      case 64 * 2432: upd2W2 = (const float*)d_in[i]; upd2W1 = (const float*)d_in[i-1]; break;
      default: break;
    }
  }

  // workspace layout
  char* ws = (char*)d_ws;
  size_t off = 0;
  auto alloc = [&](size_t bytes) -> void* {
    void* p = ws + off;
    off = (off + bytes + 255) & ~(size_t)255;
    return p;
  };
  float* ns_a  = (float*)alloc((size_t)N32 * 16 * 4);
  float* nv_a  = (float*)alloc((size_t)N * 12 * 4);
  float* ns_b  = (float*)alloc((size_t)N * 16 * 4);
  float* nv_b  = (float*)alloc((size_t)N * 12 * 4);
  float* agg_s = (float*)alloc((size_t)N * 8 * 4);
  float* agg_v = (float*)alloc((size_t)N * 6 * 4);
  _Float16* h_node = (_Float16*)alloc((size_t)N32 * 64 * 2);
  _Float16* h_edge = (_Float16*)alloc((size_t)E32 * 64 * 2);
  _Float16* pk_emb  = (_Float16*)alloc((size_t)2 * 16 * 32 * 2);
  _Float16* pk_msg1 = (_Float16*)alloc((size_t)2 * 160 * 32 * 2);
  _Float16* pk_upd1 = (_Float16*)alloc((size_t)2 * 2176 * 32 * 2);
  _Float16* pk_msg2 = (_Float16*)alloc((size_t)2 * 208 * 32 * 2);
  _Float16* pk_upd2 = (_Float16*)alloc((size_t)2 * 2432 * 32 * 2);
  _Float16* pk_intT = (_Float16*)alloc((size_t)10 * 64 * 32 * 2);
  char* BIG = (char*)alloc((size_t)50 * 1024 * 1024);  // reused scratch

  const int TPB = 256;
  auto cdiv = [](int a, int b) { return (a + b - 1) / b; };

  // zero the GEMM pad rows of the f16 activation buffers (tiny)
  if (N32 > N)
    k_zero_h<<<cdiv((N32 - N) * 64, TPB), TPB, 0, stream>>>(
        h_node + (size_t)N * 64, (N32 - N) * 64);
  if (E32 > E)
    k_zero_h<<<cdiv((E32 - E) * 64, TPB), TPB, 0, stream>>>(
        h_edge + (size_t)E * 64, (E32 - E) * 64);

  // pack all second-layer weights to WMMA fragment layout (f16)
  auto pack = [&](const float* src, _Float16* dst, int Kv, int Nv, int Npad, int tr) {
    int total = ((Kv + 31) / 32) * Npad * 32;
    if (tr) total = (320 / 32) * Npad * 32;  // inter uses Kpad=320
    k_pack_b<<<cdiv(total, TPB), TPB, 0, stream>>>(src, dst, Kv, Nv, Npad, total, tr);
  };
  pack(embW2, pk_emb, 64, 16, 16, 0);
  pack(msg1W2, pk_msg1, 64, 160, 160, 0);
  pack(upd1W2, pk_upd1, 64, 2176, 2176, 0);
  pack(msg2W2, pk_msg2, 64, 200, 208, 0);
  pack(upd2W2, pk_upd2, 64, 2432, 2432, 0);
  pack(interW2, pk_intT, 292, 64, 64, 1);

  // node embedding: ns_a = (x @ W1/sqrt(10)) @ W2/8   (no activation)
  k_row_mlp64<<<cdiv(N, TPB), TPB, 0, stream>>>(x, N, 10, embW1, 1.0f / sqrtf(10.0f), 0, h_node);
  launch_gemm<float>(h_node, pk_emb, ns_a, N32, 64, 16, 0.125f, stream);

  float* ns_cur = ns_a; float* nv_cur = nv_a;
  float* ns_new = ns_b; float* nv_new = nv_b;
  const float deg = sqrtf((float)N / (float)E);
  const int NODE_CHUNK = 3776;  // multiple of 32

  // ---------------- layer 1 ----------------
  {
    k_row_mlp64<<<cdiv(E, TPB), TPB, 0, stream>>>(ea, E, 4, msg1W1, 0.5f, 1, h_edge);
    _Float16* wm = (_Float16*)BIG;
    launch_gemm<_Float16>(h_edge, pk_msg1, wm, E32, 64, 160, 0.125f, stream);
    k_zero<<<cdiv(N * 8, TPB), TPB, 0, stream>>>(agg_s, N * 8);
    k_zero<<<cdiv(N * 6, TPB), TPB, 0, stream>>>(agg_v, N * 6);
    k_msg_scatter<false><<<cdiv(E * 8, TPB), TPB, 0, stream>>>(
        ei, E, pos, ns_cur, nv_cur, wm, 160, agg_s, agg_v, 0.25f, 0.25f, deg);
    k_row_mlp64<<<cdiv(N, TPB), TPB, 0, stream>>>(x, N, 10, upd1W1, 1.0f / sqrtf(10.0f), 1, h_node);
    _Float16* wu = (_Float16*)BIG;
    for (int base = 0; base < N; base += NODE_CHUNK) {
      int Mc = (N - base < NODE_CHUNK) ? (N - base) : NODE_CHUNK;
      int Mg = (Mc + 31) & ~31;  // padded rows exist in h_node (zeroed)
      launch_gemm<_Float16>(h_node + (size_t)base * 64, pk_upd1, wu, Mg, 64, 2176, 0.125f, stream);
      k_upd_fctp<false><<<cdiv(Mc * 16, TPB), TPB, 0, stream>>>(
          base, Mc, ns_cur, nv_cur, agg_s, agg_v, wu, 2176, ns_new, nv_new,
          1.0f / sqrtf(128.0f), 1.0f / sqrtf(32.0f));
    }
    float* t;
    t = ns_cur; ns_cur = ns_new; ns_new = t;
    t = nv_cur; nv_cur = nv_new; nv_new = t;
  }
  // ---------------- layer 2 ----------------
  {
    k_row_mlp64<<<cdiv(E, TPB), TPB, 0, stream>>>(ea, E, 4, msg2W1, 0.5f, 1, h_edge);
    _Float16* wm = (_Float16*)BIG;
    launch_gemm<_Float16>(h_edge, pk_msg2, wm, E32, 64, 208, 0.125f, stream);
    k_zero<<<cdiv(N * 8, TPB), TPB, 0, stream>>>(agg_s, N * 8);
    k_zero<<<cdiv(N * 6, TPB), TPB, 0, stream>>>(agg_v, N * 6);
    const float r20 = 1.0f / sqrtf(20.0f);
    k_msg_scatter<true><<<cdiv(E * 8, TPB), TPB, 0, stream>>>(
        ei, E, pos, ns_cur, nv_cur, wm, 208, agg_s, agg_v, r20, r20, deg);
    k_row_mlp64<<<cdiv(N, TPB), TPB, 0, stream>>>(x, N, 10, upd2W1, 1.0f / sqrtf(10.0f), 1, h_node);
    _Float16* wu = (_Float16*)BIG;
    for (int base = 0; base < N; base += NODE_CHUNK) {
      int Mc = (N - base < NODE_CHUNK) ? (N - base) : NODE_CHUNK;
      int Mg = (Mc + 31) & ~31;
      launch_gemm<_Float16>(h_node + (size_t)base * 64, pk_upd2, wu, Mg, 64, 2432, 0.125f, stream);
      k_upd_fctp<true><<<cdiv(Mc * 16, TPB), TPB, 0, stream>>>(
          base, Mc, ns_cur, nv_cur, agg_s, agg_v, wu, 2432, ns_new, nv_new,
          1.0f / sqrtf(136.0f), 0.125f);
    }
    float* t;
    t = ns_cur; ns_cur = ns_new; ns_new = t;
    t = nv_cur; nv_cur = nv_new; nv_new = t;
  }
  // ---------------- interaction head ----------------
  {
    const int IC = 37504;  // multiple of 32 (and 300000 % IC is a multiple of 32)
    _Float16* F  = (_Float16*)BIG;
    _Float16* SH = (_Float16*)(BIG + (size_t)IC * 320 * 2);
    float*    T  = (float*)(BIG + (size_t)IC * 320 * 2 + (size_t)IC * 64 * 2);
    const float tscale = 0.125f / sqrtf(292.0f);  // W2/8 and 1/sqrt(fan)
    for (int base = 0; base < Ei; base += IC) {
      int Mc = (Ei - base < IC) ? (Ei - base) : IC;
      k_inter_build<<<cdiv(Mc * 32, TPB), TPB, 0, stream>>>(
          iei, Ei, base, Mc, pos, ns_cur, nv_cur, interW1, F, SH);
      launch_gemm<float>(F, pk_intT, T, Mc, 320, 64, tscale, stream);
      k_inter_out<<<cdiv(Mc * 32, TPB), TPB, 0, stream>>>(SH, T, (float*)d_out, Mc, base);
    }
  }
  (void)out_size; (void)ws_size;
}
